// BAESNN_36378372997286
// MI455X (gfx1250) — 8-bit matrix operand order repaired and checked by reading, compiled, not run on hardware
//
#include <hip/hip_runtime.h>
#include <stdint.h>

typedef __attribute__((ext_vector_type(8))) int v8i;
union IFrag { v8i v; unsigned u[8]; };

__device__ __forceinline__ unsigned expand4(unsigned nib) {
    return (nib * 0x00204081u) & 0x01010101u;
}

__global__ void __launch_bounds__(256) snn_zero_ws(float* ws) {
    int i = threadIdx.x;
    if (i < 240) ws[i] = 0.0f;
}

struct Flags { unsigned aw, cw; };

template <bool NEED_FLAGS>
__device__ __forceinline__ Flags buildCols(const float* __restrict__ x, int row, int Bn,
                                           unsigned* __restrict__ ldsCols, int lane) {
    unsigned valid = (unsigned)__ballot(row < Bn);
    int rc = (row < Bn) ? row : (Bn - 1);
    const float4* r4 = (const float4*)(x + (size_t)rc * 40);
    unsigned aw = 0u, cw = 0u;
#pragma unroll
    for (int q = 0; q < 10; ++q) {
        float4 v = r4[q];
        unsigned c0 = (unsigned)__ballot(6.0f * v.x > 0.5f) & valid;
        unsigned c1 = (unsigned)__ballot(6.0f * v.y > 0.5f) & valid;
        unsigned c2 = (unsigned)__ballot(6.0f * v.z > 0.5f) & valid;
        unsigned c3 = (unsigned)__ballot(6.0f * v.w > 0.5f) & valid;
        if (NEED_FLAGS) {
            if (q < 5) aw |= (c0 | c1) | (c2 | c3);
            else       cw |= (c0 | c1) | (c2 | c3);
        }
        if (lane == 0) {
            uint4 pk; pk.x = c0; pk.y = c1; pk.z = c2; pk.w = c3;
            *(uint4*)(ldsCols + 4 * q) = pk;
        }
    }
    Flags f; f.aw = aw; f.cw = cw; return f;
}

__device__ __forceinline__ v8i packA(unsigned w0, unsigned w1, int sh) {
    IFrag f;
    f.u[0] = expand4((w0 >> sh)        & 0xFu);
    f.u[1] = expand4((w0 >> (sh + 4))  & 0xFu);
    f.u[2] = expand4((w0 >> (sh + 16)) & 0xFu);
    f.u[3] = expand4((w0 >> (sh + 20)) & 0xFu);
    f.u[4] = expand4((w1 >> sh)        & 0xFu);
    f.u[5] = expand4((w1 >> (sh + 4))  & 0xFu);
    f.u[6] = expand4((w1 >> (sh + 16)) & 0xFu);
    f.u[7] = expand4((w1 >> (sh + 20)) & 0xFu);
    return f.v;
}

__device__ __forceinline__ void storeRow(float* __restrict__ out, size_t OFF_IFG,
                                         size_t OFF_SMA, size_t OFF_M1,
                                         int row, unsigned a, unsigned c) {
    float av = a ? 1.0f : 0.0f;
    float cv = c ? 1.0f : 0.0f;
    float ev = (a | c) ? 1.0f : 0.0f;
    float2 aa = {av, av}, ee = {ev, ev}, cc = {cv, cv}, ac = {av, cv};
    float2* og = (float2*)(out + OFF_IFG + (size_t)row * 50);
#pragma unroll
    for (int q = 0;  q < 10; ++q) og[q] = aa;
#pragma unroll
    for (int q = 10; q < 15; ++q) og[q] = ee;
#pragma unroll
    for (int q = 15; q < 25; ++q) og[q] = cc;
    float2* os = (float2*)(out + OFF_SMA + (size_t)row * 10);
    os[0] = aa; os[1] = aa; os[2] = ac; os[3] = cc; os[4] = cc;
    float2* om = (float2*)(out + OFF_M1 + (size_t)row * 10);
    om[0] = aa; om[1] = aa; om[2] = ac; om[3] = cc; om[4] = cc;
}

__global__ void __launch_bounds__(256) snn_main(const float* __restrict__ x1,
                                                const float* __restrict__ x2,
                                                float* __restrict__ out,
                                                float* __restrict__ ws,
                                                int Bn, int nChunks) {
    __shared__ int      s_acc[240];
    __shared__ unsigned s_cols[8][4][48];
    for (int i = threadIdx.x; i < 240; i += blockDim.x) s_acc[i] = 0;
    __syncthreads();

    const int lane          = threadIdx.x & 31;
    const int waveId        = threadIdx.x >> 5;
    const int wavesPerBlock = blockDim.x >> 5;
    const int lm            = lane & 15;
    const int sh            = (lane < 16) ? 0 : 8;
    const int kb            = (lane < 16) ? 0 : 16;

    unsigned (*wc)[48] = s_cols[waveId];
    if (lane < 8) {
        wc[0][40 + lane] = 0u; wc[1][40 + lane] = 0u;
        wc[2][40 + lane] = 0u; wc[3][40 + lane] = 0u;
    }

    const size_t OFF_IFG = 4800;
    const size_t OFF_SMA = OFF_IFG + (size_t)Bn * 50;
    const size_t OFF_M1  = OFF_SMA + (size_t)Bn * 10;

    v8i accM[3] = {};
    v8i accP[3] = {};

    for (int base = blockIdx.x * wavesPerBlock; base < nChunks;
         base += gridDim.x * wavesPerBlock) {
        int chunk = base + waveId;
        int R0    = chunk * 64;
        int r0    = R0 + lane;
        int r1    = r0 + 32;
        bool cv   = (chunk < nChunks);

        Flags f0 = buildCols<true >(x1, cv ? r0 : Bn, Bn, wc[0], lane);
        Flags f1 = buildCols<true >(x1, cv ? r1 : Bn, Bn, wc[1], lane);
        (void)     buildCols<false>(x2, cv ? r0 : Bn, Bn, wc[2], lane);
        (void)     buildCols<false>(x2, cv ? r1 : Bn, Bn, wc[3], lane);

        if (cv && r0 < Bn)
            storeRow(out, OFF_IFG, OFF_SMA, OFF_M1, r0, (f0.aw >> lane) & 1u, (f0.cw >> lane) & 1u);
        if (cv && r1 < Bn)
            storeRow(out, OFF_IFG, OFF_SMA, OFF_M1, r1, (f1.aw >> lane) & 1u, (f1.cw >> lane) & 1u);

        unsigned g0 = (lm == 0) ? f0.aw : (lm == 1) ? (f0.aw | f0.cw) : (lm == 2) ? f0.cw : 0u;
        unsigned g1 = (lm == 0) ? f1.aw : (lm == 1) ? (f1.aw | f1.cw) : (lm == 2) ? f1.cw : 0u;
        IFrag Bf;
#pragma unroll
        for (int v = 0; v < 4; ++v) Bf.u[v]     = expand4((g0 >> (sh + 4*(v&1) + 16*(v>>1))) & 0xFu);
#pragma unroll
        for (int v = 0; v < 4; ++v) Bf.u[4 + v] = expand4((g1 >> (sh + 4*(v&1) + 16*(v>>1))) & 0xFu);

        unsigned w0, w1;
        w0 = wc[0][lm];      w1 = wc[1][lm];
        accM[0] = __builtin_amdgcn_wmma_i32_16x16x64_iu8(false, packA(w0, w1, sh), false, Bf.v, accM[0], false, false);
        w0 = wc[0][16 + lm]; w1 = wc[1][16 + lm];
        accM[1] = __builtin_amdgcn_wmma_i32_16x16x64_iu8(false, packA(w0, w1, sh), false, Bf.v, accM[1], false, false);
        w0 = wc[0][32 + lm]; w1 = wc[1][32 + lm];
        accM[2] = __builtin_amdgcn_wmma_i32_16x16x64_iu8(false, packA(w0, w1, sh), false, Bf.v, accM[2], false, false);

        w0 = wc[2][lm];      w1 = wc[3][lm];
        accP[0] = __builtin_amdgcn_wmma_i32_16x16x64_iu8(false, packA(w0, w1, sh), false, Bf.v, accP[0], false, false);
        w0 = wc[2][16 + lm]; w1 = wc[3][16 + lm];
        accP[1] = __builtin_amdgcn_wmma_i32_16x16x64_iu8(false, packA(w0, w1, sh), false, Bf.v, accP[1], false, false);
        w0 = wc[2][32 + lm]; w1 = wc[3][32 + lm];
        accP[2] = __builtin_amdgcn_wmma_i32_16x16x64_iu8(false, packA(w0, w1, sh), false, Bf.v, accP[2], false, false);
    }

    if (lm < 3) {
        int roff = (lane >= 16) ? 8 : 0;
#pragma unroll
        for (int t = 0; t < 3; ++t) {
#pragma unroll
            for (int r = 0; r < 8; ++r) {
                int i = 16*t + r + roff;
                if (i < 40) {
                    atomicAdd(&s_acc[i*3 + lm],       accM[t][r]);
                    atomicAdd(&s_acc[120 + i*3 + lm], accP[t][r]);
                }
            }
        }
    }
    __syncthreads();
    for (int i = threadIdx.x; i < 240; i += blockDim.x)
        atomicAdd(ws + i, (float)s_acc[i]);
}

__global__ void __launch_bounds__(256) snn_expand(const float* __restrict__ ws,
                                                  float* __restrict__ out) {
    int idx = blockIdx.x * blockDim.x + threadIdx.x;
    if (idx >= 4800) return;
    const float* dm = ws;
    const float* dp = ws + 120;
    float v;
    if (idx < 2000) {
        int i = idx / 50, j = idx % 50;
        int g = (j < 20) ? 0 : (j < 30) ? 1 : 2;
        v = dm[i*3 + g];
    } else if (idx < 4000) {
        int l = idx - 2000;
        int i = l / 50, j = l % 50;
        int g = (j < 20) ? 0 : (j < 30) ? 1 : 2;
        v = dp[i*3 + g];
    } else if (idx < 4400) {
        int l = idx - 4000;
        int i = l / 10, j = l % 10;
        v = dm[i*3 + ((j < 5) ? 0 : 2)];
    } else {
        int l = idx - 4400;
        int i = l / 10, j = l % 10;
        v = dp[i*3 + ((j < 5) ? 0 : 2)];
    }
    out[idx] = v;
}

extern "C" void kernel_launch(void* const* d_in, const int* in_sizes, int n_in,
                              void* d_out, int out_size, void* d_ws, size_t ws_size,
                              hipStream_t stream) {
    const float* x1 = (const float*)d_in[0];
    const float* x2 = (const float*)d_in[1];
    float* out = (float*)d_out;
    float* ws  = (float*)d_ws;

    int Bn      = in_sizes[0] / 40;
    int nChunks = (Bn + 63) / 64;

    snn_zero_ws<<<1, 256, 0, stream>>>(ws);

    int blocks = (nChunks + 7) / 8;
    if (blocks > 1024) blocks = 1024;
    if (blocks < 1)    blocks = 1;
    snn_main<<<blocks, 256, 0, stream>>>(x1, x2, out, ws, Bn, nChunks);

    snn_expand<<<(4800 + 255) / 256, 256, 0, stream>>>(ws, out);
}
